// MaxPoolingAggregator_89146341196167
// MI455X (gfx1250) — matrix operand order repaired and checked by reading, not compiled, not run on hardware
//
#include <hip/hip_runtime.h>
#include <math.h>

typedef __attribute__((ext_vector_type(16))) _Float16 v16h;
typedef __attribute__((ext_vector_type(8)))  float    v8f;
typedef __attribute__((ext_vector_type(4)))  float    f4;   // native vec for NT loads

#define NN   1024
#define D    64
#define BIGF 1e9f

// ---------------- pass 0: init workspace scalar ----------------
__global__ void init_ws_kernel(unsigned* ws) {
    if (threadIdx.x == 0 && blockIdx.x == 0) ws[0] = 0u;
}

// ---------------- pass 1: global max of re^2+im^2 ----------------
__global__ void __launch_bounds__(256)
max_mag2_kernel(const f4* __restrict__ re, const f4* __restrict__ im,
                long n4, unsigned* __restrict__ ws) {
    float m = 0.f;
    const long stride = (long)gridDim.x * blockDim.x;
    for (long i = (long)blockIdx.x * blockDim.x + threadIdx.x; i < n4; i += stride) {
        const f4 r = __builtin_nontemporal_load(&re[i]);
        const f4 q = __builtin_nontemporal_load(&im[i]);
        m = fmaxf(m, r.x * r.x + q.x * q.x);
        m = fmaxf(m, r.y * r.y + q.y * q.y);
        m = fmaxf(m, r.z * r.z + q.z * q.z);
        m = fmaxf(m, r.w * r.w + q.w * q.w);
    }
    __shared__ float red[256];
    red[threadIdx.x] = m;
    __syncthreads();
    for (int s = 128; s > 0; s >>= 1) {
        if (threadIdx.x < s) red[threadIdx.x] = fmaxf(red[threadIdx.x], red[threadIdx.x + s]);
        __syncthreads();
    }
    if (threadIdx.x == 0) atomicMax(ws, __float_as_uint(red[0]));
}

// pack 8 floats (two float4) into halves v[base .. base+7]
__device__ __forceinline__ void fill8(v16h& v, int base, const float4 a, const float4 b) {
    v[base + 0] = (_Float16)a.x; v[base + 1] = (_Float16)a.y;
    v[base + 2] = (_Float16)a.z; v[base + 3] = (_Float16)a.w;
    v[base + 4] = (_Float16)b.x; v[base + 5] = (_Float16)b.y;
    v[base + 6] = (_Float16)b.z; v[base + 7] = (_Float16)b.w;
}

// ---------------- pass 2: GEMM + masked argmax + select + relu ----------------
__global__ void __launch_bounds__(256)
agg_kernel(const float* __restrict__ self_re, const float* __restrict__ self_im,
           const float* __restrict__ neigh_re, const float* __restrict__ neigh_im,
           const float* __restrict__ W, const int* __restrict__ adj,
           const unsigned* __restrict__ ws, float* __restrict__ out) {
    const int n    = blockIdx.x;
    const int tid  = threadIdx.x;
    const int wave = tid >> 5;
    const int lane = tid & 31;
    const int col  = lane & 15;   // A-row / B-col / C-col position
    const int hi   = lane >> 4;   // lane half

    const float bigneg = -(sqrtf(__uint_as_float(ws[0])) + BIGF);

    // ---- preload B = W (f16), 4 channel tiles x 2 K-chunks ----
    // B fragment (16-bit, 32x16): lane holds column (lane&15), K = 16*(lane>>4)+j
    v16h B[4][2];
    #pragma unroll
    for (int t = 0; t < 4; ++t) {
        #pragma unroll
        for (int q = 0; q < 2; ++q) {
            const int kb = q * 32 + hi * 16;
            #pragma unroll
            for (int j = 0; j < 16; ++j)
                B[t][q][j] = (_Float16)W[(kb + j) * D + t * 16 + col];
        }
    }

    float bs[4], br[4], bi[4];
    #pragma unroll
    for (int t = 0; t < 4; ++t) { bs[t] = -1.f; br[t] = 0.f; bi[t] = 0.f; }

    const size_t adjrow = (size_t)n * NN;
    const float* nre = neigh_re + adjrow * D;
    const float* nim = neigh_im + adjrow * D;

    for (int it = 0; it < NN / (16 * 8); ++it) {
        const int m_base = (it * 8 + wave) * 16;

        // ---- A fragments (16-bit 16x32): lane = row (lane&15),
        //      16 contiguous K at kbase+16*hi ----
        const float* pr = nre + (size_t)(m_base + col) * D + hi * 16;
        const float* pi = nim + (size_t)(m_base + col) * D + hi * 16;
        v16h Ar[2], Ai[2];
        #pragma unroll
        for (int q = 0; q < 2; ++q) {
            const float4 r0 = *(const float4*)(pr + q * 32);
            const float4 r1 = *(const float4*)(pr + q * 32 + 4);
            const float4 r2 = *(const float4*)(pr + q * 32 + 8);
            const float4 r3 = *(const float4*)(pr + q * 32 + 12);
            fill8(Ar[q], 0, r0, r1);
            fill8(Ar[q], 8, r2, r3);
            const float4 i0 = *(const float4*)(pi + q * 32);
            const float4 i1 = *(const float4*)(pi + q * 32 + 4);
            const float4 i2 = *(const float4*)(pi + q * 32 + 8);
            const float4 i3 = *(const float4*)(pi + q * 32 + 12);
            fill8(Ai[q], 0, i0, i1);
            fill8(Ai[q], 8, i2, i3);
        }

        // ---- mask values for this lane's 8 C-rows: m = m_base + 8*hi + k ----
        const int4 a0 = *(const int4*)(adj + adjrow + m_base + hi * 8);
        const int4 a1 = *(const int4*)(adj + adjrow + m_base + hi * 8 + 4);
        float msk[8];
        msk[0] = (a0.x == 1) ? 1.f : bigneg;
        msk[1] = (a0.y == 1) ? 1.f : bigneg;
        msk[2] = (a0.z == 1) ? 1.f : bigneg;
        msk[3] = (a0.w == 1) ? 1.f : bigneg;
        msk[4] = (a1.x == 1) ? 1.f : bigneg;
        msk[5] = (a1.y == 1) ? 1.f : bigneg;
        msk[6] = (a1.z == 1) ? 1.f : bigneg;
        msk[7] = (a1.w == 1) ? 1.f : bigneg;

        #pragma unroll
        for (int t = 0; t < 4; ++t) {
            v8f cr = {}, ci = {};
            cr = __builtin_amdgcn_wmma_f32_16x16x32_f16(false, Ar[0], false, B[t][0], (short)0, cr, false, false);
            cr = __builtin_amdgcn_wmma_f32_16x16x32_f16(false, Ar[1], false, B[t][1], (short)0, cr, false, false);
            ci = __builtin_amdgcn_wmma_f32_16x16x32_f16(false, Ai[0], false, B[t][0], (short)0, ci, false, false);
            ci = __builtin_amdgcn_wmma_f32_16x16x32_f16(false, Ai[1], false, B[t][1], (short)0, ci, false, false);
            #pragma unroll
            for (int k = 0; k < 8; ++k) {
                const float fre = cr[k] * msk[k];
                const float fim = ci[k] * msk[k];
                const float s   = fre * fre + fim * fim;
                if (s > bs[t]) { bs[t] = s; br[t] = fre; bi[t] = fim; }
            }
        }
    }

    // ---- cross-lane / cross-wave reduction of (score, re, im) per channel ----
    __shared__ float sS[8][4][32], sR[8][4][32], sI[8][4][32];
    #pragma unroll
    for (int t = 0; t < 4; ++t) {
        sS[wave][t][lane] = bs[t];
        sR[wave][t][lane] = br[t];
        sI[wave][t][lane] = bi[t];
    }
    __syncthreads();

    if (tid < D) {
        const int t = tid >> 4, c = tid & 15;
        float s = -1.f, r = 0.f, q = 0.f;
        #pragma unroll
        for (int w = 0; w < 8; ++w) {
            #pragma unroll
            for (int h = 0; h < 2; ++h) {
                const float ss = sS[w][t][c + 16 * h];
                if (ss > s) { s = ss; r = sR[w][t][c + 16 * h]; q = sI[w][t][c + 16 * h]; }
            }
        }
        const size_t o = (size_t)n * D + tid;
        out[o]                    = fmaxf(self_re[o] + r, 0.f);
        out[(size_t)NN * D + o]   = fmaxf(self_im[o] + q, 0.f);
    }
}

extern "C" void kernel_launch(void* const* d_in, const int* in_sizes, int n_in,
                              void* d_out, int out_size, void* d_ws, size_t ws_size,
                              hipStream_t stream) {
    const float* self_re  = (const float*)d_in[0];
    const float* self_im  = (const float*)d_in[1];
    const float* neigh_re = (const float*)d_in[2];
    const float* neigh_im = (const float*)d_in[3];
    const float* W        = (const float*)d_in[4];
    const int*   adj      = (const int*)d_in[5];
    unsigned*    ws       = (unsigned*)d_ws;
    float*       out      = (float*)d_out;

    init_ws_kernel<<<1, 64, 0, stream>>>(ws);
    const long n4 = (long)NN * NN * D / 4;
    max_mag2_kernel<<<4096, 256, 0, stream>>>((const f4*)neigh_re,
                                              (const f4*)neigh_im, n4, ws);
    agg_kernel<<<NN, 256, 0, stream>>>(self_re, self_im, neigh_re, neigh_im,
                                       W, adj, ws, out);
}
